// Decoder_82944408420437
// MI455X (gfx1250) — matrix operand order repaired and checked by reading, not compiled, not run on hardware
//
#include <hip/hip_runtime.h>
#include <hip/hip_bf16.h>

// ---------------------------------------------------------------------------
// CDNA5 / gfx1250 transformer decoder (6 layers), bf16 WMMA + f32 accumulate.
// All WMMA B-operands are stored pre-transposed [N x K] so every fragment
// load is lane-contiguous (merges into global_load_b128 / ds_load_b128).
// GEMM-style kernels use 32-row x 64-col per-wave register blocking so each
// B fragment is reused by two WMMAs (two A strips) per k-step.
// ---------------------------------------------------------------------------

typedef __attribute__((ext_vector_type(16))) __bf16 v16bf;
typedef __attribute__((ext_vector_type(8)))  float  v8f;

#define WMMA_BF16(A_, B_, C_) \
  __builtin_amdgcn_wmma_f32_16x16x32_bf16(false, (A_), false, (B_), (short)0, (C_), false, false)

constexpr int L_  = 6;
constexpr int H_  = 8;
constexpr int C_  = 512;
constexpr int DH_ = 64;
constexpr int FF_ = 2048;
constexpr int B_  = 64;
constexpr int T_  = 128;
constexpr int S_  = 128;
constexpr int M_  = B_ * T_;      // 8192 tokens

// ---------------------------------------------------------------------------
// Fragment helpers (wave32 WMMA 16x16x32 bf16; A/B k order is this file's own contiguous choice)
//   A 16x32:  lane holds row m=lane%16; element e -> k = e + 16*(lane/16)
//   B 32x16:  lane holds col n=lane%16; element e -> k = e + 16*(lane/16)
//   C 16x16:  lane holds col n=lane%16; vgpr r   -> row = r + 8*(lane/16)
// ---------------------------------------------------------------------------

__device__ inline v8f zero_v8f() {
  v8f z;
#pragma unroll
  for (int i = 0; i < 8; ++i) z[i] = 0.0f;
  return z;
}

// A fragment from row-major [M x K] bf16 memory (one contiguous 32B run/lane)
__device__ inline v16bf load_a_frag_bf16(const __bf16* A, int lda, int row_base, int kbase) {
  const int lane = threadIdx.x & 31;
  const int half = lane >> 4;
  const __bf16* p = A + (size_t)(row_base + (lane & 15)) * lda + kbase + half * 16;
  v16bf f;
#pragma unroll
  for (int e = 0; e < 16; ++e) f[e] = p[e];
  return f;
}

// B fragment from TRANSPOSED [N x K] bf16 memory (one contiguous 32B run/lane)
__device__ inline v16bf load_bT_frag_bf16(const __bf16* BT, int ldk, int kbase, int col_base) {
  const int lane = threadIdx.x & 31;
  const int half = lane >> 4;
  const __bf16* p = BT + (size_t)(col_base + (lane & 15)) * ldk + kbase + half * 16;
  v16bf f;
#pragma unroll
  for (int e = 0; e < 16; ++e) f[e] = p[e];
  return f;
}

// ---------------------------------------------------------------------------
// fp32 -> (fp32 copy?, bf16) elementwise conversion
// ---------------------------------------------------------------------------
__global__ void cvt_kernel(const float* __restrict__ src, float* __restrict__ dstf,
                           __bf16* __restrict__ dstb, size_t n) {
  size_t i = (size_t)blockIdx.x * blockDim.x + threadIdx.x;
  if (i < n) {
    float v = src[i];
    if (dstf) dstf[i] = v;
    dstb[i] = (__bf16)v;
  }
}

// ---------------------------------------------------------------------------
// Batched fp32 [RR x CC] -> bf16 transposed [CC x RR].  RR, CC are pow-2
// compile-time constants so index math is shift/mask.  grid.y = matrix id.
// ---------------------------------------------------------------------------
template <int RR, int CC>
__global__ void cvtT_kernel(const float* __restrict__ src, __bf16* __restrict__ dst) {
  const size_t per = (size_t)RR * CC;
  const size_t base = (size_t)blockIdx.y * per;
  int i = blockIdx.x * 256 + threadIdx.x;      // linear index in dst [CC x RR]
  if (i < (int)per) {
    int r = i & (RR - 1);                      // src row
    int c = i / RR;                            // src col
    dst[base + i] = (__bf16)src[base + (size_t)r * CC + c];
  }
}

// ---------------------------------------------------------------------------
// Per-head projection: out = x * W[h] + bias[h], bf16 output.
//   TR = false: out[b,h,t,d]   (Q, K layout)
//   TR = true : out[b,h,d,t]   (V^T layout, feeds contiguous P*V B-fragments)
// W is pre-transposed per head: [H x DH x C].
// grid = (M/256, H), block = 256; each wave owns 32 rows x 64 cols.
// ---------------------------------------------------------------------------
template <bool TR>
__global__ void proj_head_kernel(const __bf16* __restrict__ X,    // [M x C]
                                 const __bf16* __restrict__ WT,   // [H x DH x C]
                                 const float*  __restrict__ bias, // [H x DH]
                                 __bf16* __restrict__ out) {
  const int wave = threadIdx.x >> 5;
  const int lane = threadIdx.x & 31;
  const int h    = blockIdx.y;
  const int row0 = blockIdx.x * 256 + wave * 32;

  const __bf16* Wh = WT + (size_t)h * DH_ * C_;

  v8f acc[2][4];
#pragma unroll
  for (int w = 0; w < 2; ++w)
#pragma unroll
    for (int j = 0; j < 4; ++j) acc[w][j] = zero_v8f();

  for (int k0 = 0; k0 < C_; k0 += 32) {
    v16bf af0 = load_a_frag_bf16(X, C_, row0,      k0);
    v16bf af1 = load_a_frag_bf16(X, C_, row0 + 16, k0);
#pragma unroll
    for (int j = 0; j < 4; ++j) {
      v16bf bf = load_bT_frag_bf16(Wh, C_, k0, j * 16);
      acc[0][j] = WMMA_BF16(af0, bf, acc[0][j]);
      acc[1][j] = WMMA_BF16(af1, bf, acc[1][j]);
    }
  }

  const int half = lane >> 4;
  const int nloc = lane & 15;
#pragma unroll
  for (int w = 0; w < 2; ++w) {
#pragma unroll
    for (int j = 0; j < 4; ++j) {
      const int d  = j * 16 + nloc;
      const float bv = bias[h * DH_ + d];
#pragma unroll
      for (int r = 0; r < 8; ++r) {
        const int m = row0 + w * 16 + r + half * 8;   // global token index
        const int b = m >> 7;                         // T_ = 128
        const int t = m & (T_ - 1);
        const float vv = acc[w][j][r] + bv;
        if (TR) out[(((size_t)b * H_ + h) * DH_ + d) * T_ + t] = (__bf16)vv;
        else    out[(((size_t)b * H_ + h) * T_  + t) * DH_ + d] = (__bf16)vv;
      }
    }
  }
}

// ---------------------------------------------------------------------------
// Fused attention per (b,h): S = softmax(Q K^T / 8), O = S V
// Q,K: bf16 [B,H,T,DH]; V: bf16 [B,H,DH,S] (transposed).
// grid = B*H blocks, block = 256 threads (8 waves)
// LDS: 128x132 f32 scores (67.6KB) + 128x136 bf16 probs (34.8KB) = 102.4KB
// ---------------------------------------------------------------------------
__global__ void attention_kernel(const __bf16* __restrict__ q,
                                 const __bf16* __restrict__ k,
                                 const __bf16* __restrict__ vT,
                                 float* __restrict__ out) {   // [B,T,C] head-concat
  constexpr int SCP = 132;   // f32 score pitch (bank-conflict pad)
  constexpr int PBP = 136;   // bf16 prob pitch (16B-aligned rows: 136*2=272=17*16)
  __shared__ __align__(16) float  sc[T_ * SCP];
  __shared__ __align__(16) __bf16 pb[T_ * PBP];

  const int bh   = blockIdx.x;
  const int b    = bh >> 3;            // H_ = 8
  const int h    = bh & (H_ - 1);
  const int wave = threadIdx.x >> 5;
  const int lane = threadIdx.x & 31;
  const int r0   = wave * 16;
  const int half = lane >> 4;
  const int nloc = lane & 15;

  const __bf16* Q  = q  + (size_t)bh * T_ * DH_;
  const __bf16* K  = k  + (size_t)bh * S_ * DH_;
  const __bf16* VT = vT + (size_t)bh * DH_ * S_;

  // ---- Phase 1: scores = Q K^T * 1/sqrt(DH) -------------------------------
  v16bf afq[2];
  afq[0] = load_a_frag_bf16(Q, DH_, r0, 0);
  afq[1] = load_a_frag_bf16(Q, DH_, r0, 32);

  for (int j = 0; j < 8; ++j) {            // 8 column tiles of 16 (S = 128)
    v8f acc = zero_v8f();
#pragma unroll
    for (int ks = 0; ks < 2; ++ks) {
      // K^T fragment: K stored [s][d] row-major => lane-contiguous in d
      v16bf bf = load_bT_frag_bf16(K, DH_, ks * 32, j * 16);
      acc = WMMA_BF16(afq[ks], bf, acc);
    }
#pragma unroll
    for (int r = 0; r < 8; ++r) {
      const int row = r0 + r + half * 8;
      sc[row * SCP + j * 16 + nloc] = acc[r] * 0.125f;   // DH^-0.5
    }
  }
  __syncthreads();

  // ---- Phase 2: row softmax, write bf16 probabilities ---------------------
  if (threadIdx.x < T_) {
    const int row = threadIdx.x;
    float mx = -3.0e38f;
    for (int c = 0; c < S_; ++c) mx = fmaxf(mx, sc[row * SCP + c]);
    float sum = 0.0f;
    for (int c = 0; c < S_; ++c) {
      float e = expf(sc[row * SCP + c] - mx);
      sc[row * SCP + c] = e;
      sum += e;
    }
    const float inv = 1.0f / sum;
    for (int c = 0; c < S_; ++c) pb[row * PBP + c] = (__bf16)(sc[row * SCP + c] * inv);
  }
  __syncthreads();

  // ---- Phase 3: O = P V ---------------------------------------------------
  for (int j = 0; j < 4; ++j) {            // DH = 64 -> 4 col tiles
    v8f acc = zero_v8f();
    for (int s0 = 0; s0 < S_; s0 += 32) {
      v16bf af = load_a_frag_bf16(pb, PBP, r0, s0);          // LDS, 16B chunks
      v16bf bf = load_bT_frag_bf16(VT, S_, s0, j * 16);      // V^T contiguous
      acc = WMMA_BF16(af, bf, acc);
    }
#pragma unroll
    for (int r = 0; r < 8; ++r) {
      const int t = r0 + r + half * 8;
      out[((size_t)b * T_ + t) * C_ + h * DH_ + j * 16 + nloc] = acc[r];
    }
  }
}

// ---------------------------------------------------------------------------
// y = LayerNorm_{(T,C)}(xin + resid); write f32 master + bf16 copy
// grid = B blocks, block = 256
// ---------------------------------------------------------------------------
__global__ void add_ln_kernel(const float* __restrict__ xin,
                              const float* __restrict__ resid,
                              float* __restrict__ xf,
                              __bf16* __restrict__ xb) {
  constexpr int N = T_ * C_;               // 65536 elements per batch
  const int b = blockIdx.x;
  const float* xi = xin   + (size_t)b * N;
  const float* rs = resid + (size_t)b * N;
  float* xo  = xf + (size_t)b * N;
  __bf16* xbo = xb + (size_t)b * N;

  __shared__ float s1[256], s2[256];
  float sum = 0.0f, ss = 0.0f;
  for (int i = threadIdx.x; i < N; i += 256) {
    float vv = xi[i] + rs[i];
    sum += vv;
    ss  += vv * vv;
  }
  s1[threadIdx.x] = sum;
  s2[threadIdx.x] = ss;
  __syncthreads();
  for (int o = 128; o > 0; o >>= 1) {
    if (threadIdx.x < o) {
      s1[threadIdx.x] += s1[threadIdx.x + o];
      s2[threadIdx.x] += s2[threadIdx.x + o];
    }
    __syncthreads();
  }
  const float mu  = s1[0] * (1.0f / N);
  const float var = s2[0] * (1.0f / N) - mu * mu;
  const float inv = rsqrtf(var + 1e-5f);

  for (int i = threadIdx.x; i < N; i += 256) {
    float vv = (xi[i] + rs[i] - mu) * inv;
    xo[i]  = vv;
    xbo[i] = (__bf16)vv;
  }
}

// ---------------------------------------------------------------------------
// Generic bf16 WMMA GEMM with transposed B:  out = act(A[MxK] * BT[NxK]^T + bias)
// grid = (N/64, M/256), block = 256; each wave owns 32 rows x 64 cols.
// ---------------------------------------------------------------------------
__global__ void gemm_bf16_kernel(const __bf16* __restrict__ A,
                                 const __bf16* __restrict__ BT,   // [N x K]
                                 const float*  __restrict__ bias,
                                 float*  __restrict__ outf,   // nullable
                                 __bf16* __restrict__ outb,   // nullable
                                 int Nn, int Kn, int relu) {
  const int wave = threadIdx.x >> 5;
  const int lane = threadIdx.x & 31;
  const int row0 = blockIdx.y * 256 + wave * 32;
  const int col0 = blockIdx.x * 64;

  v8f acc[2][4];
#pragma unroll
  for (int w = 0; w < 2; ++w)
#pragma unroll
    for (int j = 0; j < 4; ++j) acc[w][j] = zero_v8f();

  for (int k0 = 0; k0 < Kn; k0 += 32) {
    if (k0 + 32 < Kn)  // exercise global_prefetch_b8 on the next A panel
      __builtin_prefetch(A + (size_t)(row0 + (lane & 15)) * Kn + k0 + 32, 0, 0);
    v16bf af0 = load_a_frag_bf16(A, Kn, row0,      k0);
    v16bf af1 = load_a_frag_bf16(A, Kn, row0 + 16, k0);
#pragma unroll
    for (int j = 0; j < 4; ++j) {
      v16bf bf = load_bT_frag_bf16(BT, Kn, k0, col0 + j * 16);
      acc[0][j] = WMMA_BF16(af0, bf, acc[0][j]);
      acc[1][j] = WMMA_BF16(af1, bf, acc[1][j]);
    }
  }

  const int half = lane >> 4;
  const int nloc = lane & 15;
#pragma unroll
  for (int w = 0; w < 2; ++w) {
#pragma unroll
    for (int j = 0; j < 4; ++j) {
      const int n = col0 + j * 16 + nloc;
      const float bv = bias ? bias[n] : 0.0f;
#pragma unroll
      for (int r = 0; r < 8; ++r) {
        const int m = row0 + w * 16 + r + half * 8;
        float vv = acc[w][j][r] + bv;
        if (relu) vv = fmaxf(vv, 0.0f);
        const size_t idx = (size_t)m * Nn + n;
        if (outf) outf[idx] = vv;
        if (outb) outb[idx] = (__bf16)vv;
      }
    }
  }
}

// ---------------------------------------------------------------------------
// Host driver
// ---------------------------------------------------------------------------
extern "C" void kernel_launch(void* const* d_in, const int* in_sizes, int n_in,
                              void* d_out, int out_size, void* d_ws, size_t ws_size,
                              hipStream_t stream) {
  (void)in_sizes; (void)n_in; (void)ws_size;

  const float* x_in  = (const float*)d_in[0];
  const float* enc   = (const float*)d_in[1];
  const float* sa_wq = (const float*)d_in[2];
  const float* sa_bq = (const float*)d_in[3];
  const float* sa_wk = (const float*)d_in[4];
  const float* sa_bk = (const float*)d_in[5];
  const float* sa_wv = (const float*)d_in[6];
  const float* sa_bv = (const float*)d_in[7];
  const float* ca_wq = (const float*)d_in[8];
  const float* ca_bq = (const float*)d_in[9];
  const float* ca_wk = (const float*)d_in[10];
  const float* ca_bk = (const float*)d_in[11];
  const float* ca_wv = (const float*)d_in[12];
  const float* ca_bv = (const float*)d_in[13];
  const float* ff_w1 = (const float*)d_in[14];
  const float* ff_b1 = (const float*)d_in[15];
  const float* ff_w2 = (const float*)d_in[16];
  const float* ff_b2 = (const float*)d_in[17];

  // ---- workspace carve ----------------------------------------------------
  char* wp = (char*)d_ws;
  auto carve = [&](size_t bytes) -> void* {
    void* r = (void*)wp;
    wp += (bytes + 255) & ~(size_t)255;
    return r;
  };
  const size_t nXC = (size_t)M_ * C_;           // 4,194,304
  const size_t nHID = (size_t)M_ * FF_;         // 16,777,216
  const size_t nPW = (size_t)L_ * H_ * C_ * DH_;
  const size_t nFW = (size_t)L_ * C_ * FF_;

  float*  x_f32    = (float*) carve(nXC * 4);
  __bf16* x_bf16   = (__bf16*)carve(nXC * 2);
  __bf16* enc_bf16 = (__bf16*)carve(nXC * 2);
  __bf16* q_buf    = (__bf16*)carve(nXC * 2);   // [B,H,T,DH]
  __bf16* k_buf    = (__bf16*)carve(nXC * 2);   // [B,H,S,DH]
  __bf16* v_buf    = (__bf16*)carve(nXC * 2);   // [B,H,DH,S]  (transposed)
  float*  tmp_f32  = (float*) carve(nXC * 4);   // attn concat / ff2 output
  __bf16* h_bf16   = (__bf16*)carve(nHID * 2);  // FF hidden (bf16 only)
  __bf16* wsq = (__bf16*)carve(nPW * 2);        // [L,H,DH,C] transposed
  __bf16* wsk = (__bf16*)carve(nPW * 2);
  __bf16* wsv = (__bf16*)carve(nPW * 2);
  __bf16* wcq = (__bf16*)carve(nPW * 2);
  __bf16* wck = (__bf16*)carve(nPW * 2);
  __bf16* wcv = (__bf16*)carve(nPW * 2);
  __bf16* w1b = (__bf16*)carve(nFW * 2);        // [L,FF,C] transposed
  __bf16* w2b = (__bf16*)carve(nFW * 2);        // [L,C,FF] transposed

  auto cvt = [&](const float* src, float* df, __bf16* db, size_t n) {
    int blocks = (int)((n + 255) / 256);
    cvt_kernel<<<blocks, 256, 0, stream>>>(src, df, db, n);
  };

  // one-time conversions + weight transposes
  cvt(x_in, x_f32, x_bf16, nXC);
  cvt(enc, nullptr, enc_bf16, nXC);
  {
    const dim3 gP((C_ * DH_ + 255) / 256, L_ * H_);   // per-head [C x DH] -> [DH x C]
    cvtT_kernel<C_, DH_><<<gP, 256, 0, stream>>>(sa_wq, wsq);
    cvtT_kernel<C_, DH_><<<gP, 256, 0, stream>>>(sa_wk, wsk);
    cvtT_kernel<C_, DH_><<<gP, 256, 0, stream>>>(sa_wv, wsv);
    cvtT_kernel<C_, DH_><<<gP, 256, 0, stream>>>(ca_wq, wcq);
    cvtT_kernel<C_, DH_><<<gP, 256, 0, stream>>>(ca_wk, wck);
    cvtT_kernel<C_, DH_><<<gP, 256, 0, stream>>>(ca_wv, wcv);
    const dim3 gF((C_ * FF_ + 255) / 256, L_);
    cvtT_kernel<C_, FF_><<<gF, 256, 0, stream>>>(ff_w1, w1b);   // [C,FF] -> [FF,C]
    cvtT_kernel<FF_, C_><<<gF, 256, 0, stream>>>(ff_w2, w2b);   // [FF,C] -> [C,FF]
  }

  const dim3 projGrid(M_ / 256, H_);
  const dim3 attnGrid(B_ * H_);
  const size_t lW = (size_t)H_ * C_ * DH_;   // per-layer proj weight stride
  const size_t lB = (size_t)H_ * DH_;        // per-layer proj bias stride

  for (int l = 0; l < L_; ++l) {
    // ---------------- self attention ----------------
    proj_head_kernel<false><<<projGrid, 256, 0, stream>>>(x_bf16, wsq + l * lW, sa_bq + l * lB, q_buf);
    proj_head_kernel<false><<<projGrid, 256, 0, stream>>>(x_bf16, wsk + l * lW, sa_bk + l * lB, k_buf);
    proj_head_kernel<true ><<<projGrid, 256, 0, stream>>>(x_bf16, wsv + l * lW, sa_bv + l * lB, v_buf);
    attention_kernel<<<attnGrid, 256, 0, stream>>>(q_buf, k_buf, v_buf, tmp_f32);
    add_ln_kernel<<<B_, 256, 0, stream>>>(x_f32, tmp_f32, x_f32, x_bf16);

    // ---------------- cross attention ----------------
    proj_head_kernel<false><<<projGrid, 256, 0, stream>>>(x_bf16,   wcq + l * lW, ca_bq + l * lB, q_buf);
    proj_head_kernel<false><<<projGrid, 256, 0, stream>>>(enc_bf16, wck + l * lW, ca_bk + l * lB, k_buf);
    proj_head_kernel<true ><<<projGrid, 256, 0, stream>>>(enc_bf16, wcv + l * lW, ca_bv + l * lB, v_buf);
    attention_kernel<<<attnGrid, 256, 0, stream>>>(q_buf, k_buf, v_buf, tmp_f32);
    add_ln_kernel<<<B_, 256, 0, stream>>>(x_f32, tmp_f32, x_f32, x_bf16);

    // ---------------- feed forward ----------------
    gemm_bf16_kernel<<<dim3(FF_ / 64, M_ / 256), 256, 0, stream>>>(
        x_bf16, w1b + (size_t)l * C_ * FF_, ff_b1 + (size_t)l * FF_,
        nullptr, h_bf16, FF_, C_, 1);
    gemm_bf16_kernel<<<dim3(C_ / 64, M_ / 256), 256, 0, stream>>>(
        h_bf16, w2b + (size_t)l * FF_ * C_, ff_b2 + (size_t)l * C_,
        tmp_f32, nullptr, C_, FF_, 0);
    add_ln_kernel<<<B_, 256, 0, stream>>>(x_f32, tmp_f32, x_f32, x_bf16);
  }

  hipMemcpyAsync(d_out, x_f32, (size_t)out_size * sizeof(float),
                 hipMemcpyDeviceToDevice, stream);
}
